// TreeNNBatch_84061099917534
// MI455X (gfx1250) — matrix operand order repaired and checked by reading, compiled, not run on hardware
//
#include <hip/hip_runtime.h>
#include <hip/hip_bf16.h>
#include <math.h>

// ---------------- types ----------------
typedef __attribute__((ext_vector_type(16))) _Float16 v16h;
typedef __attribute__((ext_vector_type(8)))  _Float16 v8h;
typedef __attribute__((ext_vector_type(8)))  float    v8f;
typedef __attribute__((ext_vector_type(4)))  float    v4f;

// ---------------- problem constants ----------------
constexpr int Bc     = 2048;
constexpr int NODES  = 63;
constexpr int OPd    = 32;
constexpr int PREDd  = 128;
constexpr int HIDd   = 128;
constexpr int FEATd  = 64;
constexpr int BMd    = 1000;
constexpr int BMpad  = 1024;
constexpr int XCOLS  = 896;   // 5*128 + 256
constexpr int RALL   = Bc * NODES;       // 129024 rows

constexpr int FLAG_RELU    = 1;
constexpr int FLAG_PAIRMIN = 2;

// ---------------- device helpers ----------------
__device__ __forceinline__ v8h ld8(const _Float16* p) {
    return *(const v8h*)p;
}
__device__ __forceinline__ v8h ld8(const float* p) {
    v4f a = *(const v4f*)p;
    v4f b = *(const v4f*)(p + 4);
    v8h r;
#pragma unroll
    for (int i = 0; i < 4; ++i) { r[i] = (_Float16)a[i]; r[4 + i] = (_Float16)b[i]; }
    return r;
}

template <typename AT>
__device__ __forceinline__ v8h ld8_masked(const AT* row, int ks, int K) {
    if (ks + 8 <= K) return ld8(row + ks);
    v8h r = {};
#pragma unroll
    for (int i = 0; i < 8; ++i)
        if (ks + i < K) r[i] = (_Float16)row[ks + i];
    return r;
}

__device__ __forceinline__ v16h cat8(v8h lo, v8h hi) {
    return __builtin_shufflevector(lo, hi, 0,1,2,3,4,5,6,7,8,9,10,11,12,13,14,15);
}

// A fragment (16x32 f16): lanes 0-15: M=lane, K = k0+[0..7] and k0+[16..23]
//                         lanes 16-31: M=lane-16, K = k0+[8..15] and k0+[24..31]
template <typename AT>
__device__ __forceinline__ v16h load_a_frag(const AT* arow, int k0, int koff) {
    return cat8(ld8(arow + k0 + koff), ld8(arow + k0 + koff + 16));
}
template <typename AT>
__device__ __forceinline__ v16h load_a_frag_tail(const AT* arow, int k0, int koff, int K) {
    return cat8(ld8_masked(arow, k0 + koff, K), ld8_masked(arow, k0 + koff + 16, K));
}

// B fragment (32x16 f16), Wt stored transposed (N, Kpad);
// lanes 0-15: N=lane, K=k0+[0..7] and k0+[16..23]; lanes 16-31: N=lane-16, K=k0+[8..15] and k0+[24..31]
__device__ __forceinline__ v16h load_b_frag(const _Float16* brow, int k0) {
    int kh = (threadIdx.x & 16) ? 8 : 0;
    return cat8(*(const v8h*)(brow + k0 + kh), *(const v8h*)(brow + k0 + kh + 16));
}

#define WMMA_F16(a, b, c) \
    __builtin_amdgcn_wmma_f32_16x16x32_f16(false, (a), false, (b), (short)0, (c), false, false)

// ---------------- generic WMMA GEMM ----------------
// wave tile: 32 (M) x 64 (N) -> 8x v_wmma_f32_16x16x32_f16 per k-chunk.
// All 6 fragments of a chunk are loaded into distinct registers before the
// WMMAs so the scheduler can overlap next-chunk loads with current WMMAs.
template <typename AT>
__global__ void gemm32x64(const AT* __restrict__ A, const _Float16* __restrict__ Bt,
                          const float* __restrict__ bias, _Float16* __restrict__ out,
                          const float* __restrict__ rowscale,
                          int M, int N, int K, int Kpad, int lda, int ldc,
                          int npg, int gstride, int abase, int colbase, int flags) {
    int wave = (blockIdx.x * blockDim.x + threadIdx.x) >> 5;
    int lane = threadIdx.x & 31;
    int tilesN = N >> 6;
    int tm = wave / tilesN;
    int tn = wave - tm * tilesN;
    int m0 = tm << 5;
    if (m0 >= M) return;
    int n0 = tn << 6;

    int koff = (lane < 16) ? 0 : 8;      // A K sub-offset by lane half

    // per-lane A rows for the two 16-row fragments (row remap for tree slices)
    const AT* Ap0;
    const AT* Ap1;
    {
        int lrow0 = m0 + (lane & 15);
        int lrow1 = lrow0 + 16;
        int arow0 = (lrow0 / npg) * gstride + abase + (lrow0 % npg);
        int arow1 = (lrow1 / npg) * gstride + abase + (lrow1 % npg);
        Ap0 = A + (size_t)arow0 * lda;
        Ap1 = A + (size_t)arow1 * lda;
    }
    // four B row pointers (one per 16-col subtile), hoisted out of the loop
    const _Float16* Bp0 = Bt + (size_t)(n0 + (lane & 15)) * Kpad;
    const _Float16* Bp1 = Bp0 + (size_t)16 * Kpad;
    const _Float16* Bp2 = Bp0 + (size_t)32 * Kpad;
    const _Float16* Bp3 = Bp0 + (size_t)48 * Kpad;

    v8f acc[2][4] = {};

    int kfull = K - (K & 31);            // full 32-chunks
    for (int k0 = 0; k0 < kfull; k0 += 32) {
        v16h a0 = load_a_frag(Ap0, k0, koff);
        v16h a1 = load_a_frag(Ap1, k0, koff);
        v16h b0 = load_b_frag(Bp0, k0);
        v16h b1 = load_b_frag(Bp1, k0);
        v16h b2 = load_b_frag(Bp2, k0);
        v16h b3 = load_b_frag(Bp3, k0);
        acc[0][0] = WMMA_F16(a0, b0, acc[0][0]);
        acc[1][0] = WMMA_F16(a1, b0, acc[1][0]);
        acc[0][1] = WMMA_F16(a0, b1, acc[0][1]);
        acc[1][1] = WMMA_F16(a1, b1, acc[1][1]);
        acc[0][2] = WMMA_F16(a0, b2, acc[0][2]);
        acc[1][2] = WMMA_F16(a1, b2, acc[1][2]);
        acc[0][3] = WMMA_F16(a0, b3, acc[0][3]);
        acc[1][3] = WMMA_F16(a1, b3, acc[1][3]);
    }
    if (kfull < K) {                     // masked tail (Bt zero-padded to Kpad)
        v16h a0 = load_a_frag_tail(Ap0, kfull, koff, K);
        v16h a1 = load_a_frag_tail(Ap1, kfull, koff, K);
        v16h b0 = load_b_frag(Bp0, kfull);
        v16h b1 = load_b_frag(Bp1, kfull);
        v16h b2 = load_b_frag(Bp2, kfull);
        v16h b3 = load_b_frag(Bp3, kfull);
        acc[0][0] = WMMA_F16(a0, b0, acc[0][0]);
        acc[1][0] = WMMA_F16(a1, b0, acc[1][0]);
        acc[0][1] = WMMA_F16(a0, b1, acc[0][1]);
        acc[1][1] = WMMA_F16(a1, b1, acc[1][1]);
        acc[0][2] = WMMA_F16(a0, b2, acc[0][2]);
        acc[1][2] = WMMA_F16(a1, b2, acc[1][2]);
        acc[0][3] = WMMA_F16(a0, b3, acc[0][3]);
        acc[1][3] = WMMA_F16(a1, b3, acc[1][3]);
    }

    bool relu    = (flags & FLAG_RELU)    != 0;
    bool pairmin = (flags & FLAG_PAIRMIN) != 0;

#pragma unroll
    for (int i = 0; i < 2; ++i) {
        int rbase = m0 + i * 16 + ((lane < 16) ? 0 : 8);
#pragma unroll
        for (int j = 0; j < 4; ++j) {
            int col = n0 + j * 16 + (lane & 15);
            float bs = bias ? bias[col] : 0.0f;
            if (!pairmin) {
#pragma unroll
                for (int r = 0; r < 8; ++r) {
                    float v = acc[i][j][r] + bs;
                    if (rowscale) v *= rowscale[rbase + r];
                    if (relu) v = fmaxf(v, 0.0f);
                    out[(size_t)(rbase + r) * ldc + colbase + col] = (_Float16)v;
                }
            } else {
                // min over adjacent row pairs (K=2 predicate reduction)
                int obase = ((m0 + i * 16) >> 1) + ((lane < 16) ? 0 : 4);
#pragma unroll
                for (int rr = 0; rr < 4; ++rr) {
                    float v = fminf(acc[i][j][2 * rr], acc[i][j][2 * rr + 1]) + bs;
                    out[(size_t)(obase + rr) * ldc + colbase + col] = (_Float16)v;
                }
            }
        }
    }
}

// ---------------- prologue / glue kernels ----------------
// transpose + cvt: src f32 (K,N) -> dst f16 (N,Kpad), zero pad K..Kpad
__global__ void wtrans(const float* __restrict__ src, _Float16* __restrict__ dst,
                       int K, int N, int Kpad) {
    int idx = blockIdx.x * blockDim.x + threadIdx.x;
    if (idx >= N * Kpad) return;
    int n = idx / Kpad, kp = idx - n * Kpad;
    dst[idx] = (kp < K) ? (_Float16)src[(size_t)kp * N + n] : (_Float16)0.0f;
}

// zero left/right slots (cols 640..895) for leaf rows (nodes 31..62)
__global__ void zero_leaf(_Float16* __restrict__ X) {
    int idx = blockIdx.x * blockDim.x + threadIdx.x;
    if (idx >= Bc * 32 * 32) return;            // 8 halves per thread
    int c8 = idx & 31;                          // which 8-elem group of 256
    int t = idx >> 5;
    int node = 31 + (t & 31);
    int b = t >> 5;
    v8h z = {};
    *(v8h*)(X + (size_t)(b * NODES + node) * XCOLS + 640 + c8 * 8) = z;
}

// scatter child outputs C (B*n,128) into parents' left/right slots of X
__global__ void scatter_child(const _Float16* __restrict__ C, _Float16* __restrict__ X,
                              int n, int s) {
    int idx = blockIdx.x * blockDim.x + threadIdx.x;
    if (idx >= Bc * n * 16) return;             // 8 halves per thread
    int c8 = idx & 15;
    int r = idx >> 4;
    int j = r % n;
    int b = r / n;
    int node = s + j;
    int parent = (node - 1) >> 1;
    int off = (node & 1) ? 640 : 768;           // odd node index = left child
    *(v8h*)(X + (size_t)(b * NODES + parent) * XCOLS + off + c8 * 8) =
        *(const v8h*)(C + (size_t)r * 128 + c8 * 8);
}

// heads: sigmoid(dot(h,W)+b) -> out[0:B]=cost, out[B:2B]=card
__global__ void head_out(const _Float16* __restrict__ Hc, const _Float16* __restrict__ Hk,
                         const float* __restrict__ Wco, const float* __restrict__ bco,
                         const float* __restrict__ Wko, const float* __restrict__ bko,
                         float* __restrict__ out) {
    int r = blockIdx.x * blockDim.x + threadIdx.x;
    if (r >= Bc) return;
    float sc = 0.0f, sk = 0.0f;
#pragma unroll 8
    for (int i = 0; i < 128; ++i) {
        sc += (float)Hc[(size_t)r * 128 + i] * Wco[i];
        sk += (float)Hk[(size_t)r * 128 + i] * Wko[i];
    }
    out[r]      = 1.0f / (1.0f + expf(-(sc + bco[0])));
    out[Bc + r] = 1.0f / (1.0f + expf(-(sk + bko[0])));
}

// ---------------- host side ----------------
static inline void run_gemm(hipStream_t s, const void* A, bool a_half,
                            const _Float16* Bt, const float* bias, _Float16* out,
                            const float* rsc, int M, int N, int K, int Kpad,
                            int lda, int ldc, int npg, int gstride, int abase,
                            int colbase, int flags) {
    int waves  = (M / 32) * (N / 64);
    int blocks = (waves + 7) / 8;     // 256 threads = 8 waves/block
    if (a_half)
        gemm32x64<_Float16><<<blocks, 256, 0, s>>>((const _Float16*)A, Bt, bias, out,
            rsc, M, N, K, Kpad, lda, ldc, npg, gstride, abase, colbase, flags);
    else
        gemm32x64<float><<<blocks, 256, 0, s>>>((const float*)A, Bt, bias, out,
            rsc, M, N, K, Kpad, lda, ldc, npg, gstride, abase, colbase, flags);
}

extern "C" void kernel_launch(void* const* d_in, const int* in_sizes, int n_in,
                              void* d_out, int out_size, void* d_ws, size_t ws_size,
                              hipStream_t stream) {
    // inputs (setup_inputs dict order)
    const float* ops      = (const float*)d_in[0];
    const float* feats    = (const float*)d_in[1];
    const float* bitmaps  = (const float*)d_in[2];
    const float* preds1   = (const float*)d_in[3];
    const float* preds2   = (const float*)d_in[4];
    const float* has_cond = (const float*)d_in[5];
    const float* W_op   = (const float*)d_in[6];   const float* b_op   = (const float*)d_in[7];
    const float* W_pred = (const float*)d_in[8];   const float* b_pred = (const float*)d_in[9];
    const float* W_bm   = (const float*)d_in[10];  const float* b_bm   = (const float*)d_in[11];
    const float* W_feat = (const float*)d_in[12];  const float* b_feat = (const float*)d_in[13];
    const float* W_r1   = (const float*)d_in[14];  const float* b_r1   = (const float*)d_in[15];
    const float* W_r2   = (const float*)d_in[16];  const float* b_r2   = (const float*)d_in[17];
    const float* W_r3   = (const float*)d_in[18];  const float* b_r3   = (const float*)d_in[19];
    const float* W_c1   = (const float*)d_in[20];  const float* b_c1   = (const float*)d_in[21];
    const float* W_c2   = (const float*)d_in[22];  const float* b_c2   = (const float*)d_in[23];
    const float* W_co   = (const float*)d_in[24];  const float* b_co   = (const float*)d_in[25];
    const float* W_k1   = (const float*)d_in[26];  const float* b_k1   = (const float*)d_in[27];
    const float* W_k2   = (const float*)d_in[28];  const float* b_k2   = (const float*)d_in[29];
    const float* W_ko   = (const float*)d_in[30];  const float* b_ko   = (const float*)d_in[31];

    // workspace carve-out (all f16)
    char* w = (char*)d_ws;
    auto take = [&](size_t elems) -> _Float16* {
        _Float16* p = (_Float16*)w;
        w += ((elems * sizeof(_Float16) + 255) / 256) * 256;
        return p;
    };
    _Float16* Wt_op   = take((size_t)HIDd * OPd);
    _Float16* Wt_feat = take((size_t)HIDd * FEATd);
    _Float16* Wt_pred = take((size_t)HIDd * PREDd);
    _Float16* Wt_bm   = take((size_t)HIDd * BMpad);
    _Float16* Wt_r1   = take((size_t)512 * XCOLS);
    _Float16* Wt_r2   = take((size_t)256 * 512);
    _Float16* Wt_r3   = take((size_t)128 * 256);
    _Float16* Wt_c1   = take((size_t)128 * 128);
    _Float16* Wt_c2   = take((size_t)128 * 128);
    _Float16* Wt_k1   = take((size_t)128 * 128);
    _Float16* Wt_k2   = take((size_t)128 * 128);
    _Float16* X       = take((size_t)RALL * XCOLS);        // concatenated inputs
    _Float16* h1      = take((size_t)Bc * 32 * 512);
    _Float16* h2      = take((size_t)Bc * 32 * 256);
    _Float16* Cb      = take((size_t)Bc * 32 * 128);       // per-level child out / root
    _Float16* hc1     = take((size_t)Bc * 128);
    _Float16* hc2     = take((size_t)Bc * 128);
    _Float16* hk1     = take((size_t)Bc * 128);
    _Float16* hk2     = take((size_t)Bc * 128);
    (void)ws_size; (void)in_sizes; (void)n_in; (void)out_size;

    auto tw = [&](const float* src, _Float16* dst, int K, int N, int Kpad) {
        int tot = N * Kpad;
        wtrans<<<(tot + 255) / 256, 256, 0, stream>>>(src, dst, K, N, Kpad);
    };
    // weight transpose + f16 convert (+ K padding for bitmaps)
    tw(W_op,   Wt_op,   OPd,   HIDd, OPd);
    tw(W_feat, Wt_feat, FEATd, HIDd, FEATd);
    tw(W_pred, Wt_pred, PREDd, HIDd, PREDd);
    tw(W_bm,   Wt_bm,   BMd,   HIDd, BMpad);
    tw(W_r1,   Wt_r1,   XCOLS, 512,  XCOLS);
    tw(W_r2,   Wt_r2,   512,   256,  512);
    tw(W_r3,   Wt_r3,   256,   128,  256);
    tw(W_c1,   Wt_c1,   128,   128,  128);
    tw(W_c2,   Wt_c2,   128,   128,  128);
    tw(W_k1,   Wt_k1,   128,   128,  128);
    tw(W_k2,   Wt_k2,   128,   128,  128);

    // leaves have zero children
    zero_leaf<<<(Bc * 32 * 32 + 255) / 256, 256, 0, stream>>>(X);

    // ---- embeddings -> X columns [op|feat|cond1|cond2|bm] ----
    run_gemm(stream, ops,     false, Wt_op,   b_op,   X, nullptr,
             RALL, HIDd, OPd, OPd, OPd, XCOLS, RALL, 0, 0, /*col*/0,   0);
    run_gemm(stream, feats,   false, Wt_feat, b_feat, X, nullptr,
             RALL, HIDd, FEATd, FEATd, FEATd, XCOLS, RALL, 0, 0, /*col*/128, 0);
    run_gemm(stream, preds1,  false, Wt_pred, b_pred, X, nullptr,
             RALL * 2, HIDd, PREDd, PREDd, PREDd, XCOLS, RALL * 2, 0, 0, /*col*/256,
             FLAG_PAIRMIN);
    run_gemm(stream, preds2,  false, Wt_pred, b_pred, X, nullptr,
             RALL * 2, HIDd, PREDd, PREDd, PREDd, XCOLS, RALL * 2, 0, 0, /*col*/384,
             FLAG_PAIRMIN);
    run_gemm(stream, bitmaps, false, Wt_bm,   b_bm,   X, has_cond,
             RALL, HIDd, BMd, BMpad, BMd, XCOLS, RALL, 0, 0, /*col*/512, 0);

    // ---- tree MLP, leaves -> root ----
    for (int l = 5; l >= 0; --l) {
        int n = 1 << l;
        int s = n - 1;
        int M = Bc * n;
        run_gemm(stream, X,  true, Wt_r1, b_r1, h1, nullptr,
                 M, 512, XCOLS, XCOLS, XCOLS, 512, n, NODES, s, 0, FLAG_RELU);
        run_gemm(stream, h1, true, Wt_r2, b_r2, h2, nullptr,
                 M, 256, 512, 512, 512, 256, M, 0, 0, 0, FLAG_RELU);
        run_gemm(stream, h2, true, Wt_r3, b_r3, Cb, nullptr,
                 M, 128, 256, 256, 256, 128, M, 0, 0, 0, FLAG_RELU);
        if (l > 0)
            scatter_child<<<(Bc * n * 16 + 255) / 256, 256, 0, stream>>>(Cb, X, n, s);
    }
    // Cb now holds root (B,128) in f16

    // ---- heads ----
    run_gemm(stream, Cb,  true, Wt_c1, b_c1, hc1, nullptr,
             Bc, 128, 128, 128, 128, 128, Bc, 0, 0, 0, FLAG_RELU);
    run_gemm(stream, hc1, true, Wt_c2, b_c2, hc2, nullptr,
             Bc, 128, 128, 128, 128, 128, Bc, 0, 0, 0, FLAG_RELU);
    run_gemm(stream, Cb,  true, Wt_k1, b_k1, hk1, nullptr,
             Bc, 128, 128, 128, 128, 128, Bc, 0, 0, 0, FLAG_RELU);
    run_gemm(stream, hk1, true, Wt_k2, b_k2, hk2, nullptr,
             Bc, 128, 128, 128, 128, 128, Bc, 0, 0, 0, FLAG_RELU);

    head_out<<<(Bc + 255) / 256, 256, 0, stream>>>(hc2, hk2, W_co, b_co, W_ko, b_ko,
                                                   (float*)d_out);
}
